// XAModel_3264175145544
// MI455X (gfx1250) — matrix operand order repaired and checked by reading, not compiled, not run on hardware
//
#include <hip/hip_runtime.h>
#include <cmath>

#define BB   16
#define SS   1024
#define DD   64
#define HH   4
#define LL   2
#define DFFN 256
#define MM   (BB * SS)

typedef __attribute__((ext_vector_type(16))) __bf16 v16bf;
typedef __attribute__((ext_vector_type(8)))  __bf16 v8bf;
typedef __attribute__((ext_vector_type(8)))  float  v8f;

// ---- optional gfx1250 async global->LDS path (guarded, with fallback) ------
#if defined(__has_builtin)
#if __has_builtin(__builtin_amdgcn_global_load_async_to_lds_b128) && \
    __has_builtin(__builtin_amdgcn_s_wait_asynccnt)
#define ASYNC_EN 1
#endif
#endif
#ifndef ASYNC_EN
#define ASYNC_EN 0
#endif

#if ASYNC_EN
typedef int v4i __attribute__((vector_size(16)));
typedef __attribute__((address_space(1))) v4i gv4i;   // global int4
typedef __attribute__((address_space(3))) v4i lv4i;   // LDS int4
static __device__ inline void cp16_async(const __bf16* g, __bf16* l) {
  __builtin_amdgcn_global_load_async_to_lds_b128((gv4i*)g, (lv4i*)l, 0, 0);
}
#endif

// ---------------------------------------------------------------------------
// XOR-lane exchange via DS_SWIZZLE_B32 (group-of-32 mode: and=0x1F, xor<<10).
// Single DS op per step, no bpermute address math.
// ---------------------------------------------------------------------------
template <int PAT>
static __device__ inline float ds_swz(float v) {
  return __int_as_float(__builtin_amdgcn_ds_swizzle(__float_as_int(v), PAT));
}
// reduce across 16-lane halves (xor 1,2,4,8)
static __device__ inline float half_max(float v) {
  v = fmaxf(v, ds_swz<0x041F>(v));
  v = fmaxf(v, ds_swz<0x081F>(v));
  v = fmaxf(v, ds_swz<0x101F>(v));
  v = fmaxf(v, ds_swz<0x201F>(v));
  return v;
}
static __device__ inline float half_sum(float v) {
  v += ds_swz<0x041F>(v);
  v += ds_swz<0x081F>(v);
  v += ds_swz<0x101F>(v);
  v += ds_swz<0x201F>(v);
  return v;
}
// reduce across the full wave32 (xor 1,2,4,8,16)
static __device__ inline float wave_sum(float v) {
  v += ds_swz<0x041F>(v);
  v += ds_swz<0x081F>(v);
  v += ds_swz<0x101F>(v);
  v += ds_swz<0x201F>(v);
  v += ds_swz<0x401F>(v);
  return v;
}

// ---------------------------------------------------------------------------
// WMMA 16x16x32 bf16 fragment layouts (ISA 7.12.2):
// A (16x32 MxK): lanes 0-15 M=lane, K = {kb..kb+7, 16+kb..16+kb+7}, kb=0;
//                lanes 16-31 same with kb=8.
// B (32x16 KxN): N=lane%16, K split as A. This code feeds A and B K=16h..16h+15 (h=lane/16).
// C (16x16):     VGPR r: lanes 0-15 -> (M=r, N=lane); lanes 16-31 -> M=8+r.
// ---------------------------------------------------------------------------
static __device__ inline v16bf cat8(v8bf lo, v8bf hi) {
  v16bf r;
#pragma unroll
  for (int i = 0; i < 8; ++i) { r[i] = lo[i]; r[8 + i] = hi[i]; }
  return r;
}

// ---------------------------------------------------------------------------
// Weight prep: W[K,N] f32 -> Wt[N,K] bf16 (contiguous-K B fragments).
// ---------------------------------------------------------------------------
__global__ __launch_bounds__(64) void wt_cvt_t(const float* __restrict__ W,
                                               __bf16* __restrict__ Wt,
                                               int K, int N) {
  const int n = blockIdx.x;
  for (int k = threadIdx.x; k < K; k += 64)
    Wt[(size_t)n * K + k] = (__bf16)W[(size_t)k * N + n];
}

// ---------------------------------------------------------------------------
// GEMM: C[M,N] = act(A[M,K] @ Wt^T + bias). A bf16 row-major, Wt bf16 [N,K].
// OUTMODE: 0 = f32 row-major, 1 = bf16 row-major, 2 = bf16 V^T ((b,h) depth-major)
// ---------------------------------------------------------------------------
template <int K, int N, int OUTMODE, int RELU>
__global__ __launch_bounds__(32) void gemm_bf16t(const __bf16* __restrict__ A,
                                                 const __bf16* __restrict__ Wt,
                                                 const float* __restrict__ bias,
                                                 float* __restrict__ Cf,
                                                 __bf16* __restrict__ Cb) {
  const int tm = blockIdx.x * 16;
  const int tn = blockIdx.y * 16;
  const int lane = threadIdx.x & 31;
  const int m = lane & 15;
  const int kb = (lane < 16) ? 0 : 8;
  const int kb16 = (lane < 16) ? 0 : 16;
  const __bf16* arow = A + (size_t)(tm + m) * K;
  const __bf16* brow = Wt + (size_t)(tn + (lane & 15)) * K;
  v8f acc = {};
#pragma unroll
  for (int k0 = 0; k0 < K; k0 += 32) {
    v8bf alo = *(const v8bf*)(arow + k0 + kb16);
    v8bf ahi = *(const v8bf*)(arow + k0 + kb16 + 8);
    v8bf blo = *(const v8bf*)(brow + k0 + kb16);
    v8bf bhi = *(const v8bf*)(brow + k0 + kb16 + 8);
    acc = __builtin_amdgcn_wmma_f32_16x16x32_bf16(false, cat8(alo, ahi), false,
                                                  cat8(blo, bhi), (short)0, acc,
                                                  false, false);
  }
  const int ncol = tn + (lane & 15);
  const int rb = (lane < 16) ? 0 : 8;
  const float bv = bias ? bias[ncol] : 0.f;
#pragma unroll
  for (int r = 0; r < 8; ++r) {
    float v = acc[r] + bv;
    if (RELU) v = fmaxf(v, 0.f);
    const int row = tm + rb + r;
    if (OUTMODE == 0) {
      Cf[(size_t)row * N + ncol] = v;
    } else if (OUTMODE == 1) {
      Cb[(size_t)row * N + ncol] = (__bf16)v;
    } else {  // V^T: row = b*S+s, ncol = h*16+d -> Vt[((b*H+h)*16+d)*S + s]
      const int bidx = row >> 10;
      const int s = row & 1023;
      const int h = ncol >> 4, d = ncol & 15;
      Cb[(size_t)(((bidx * HH + h) * 16) + d) * SS + s] = (__bf16)v;
    }
  }
}

// ---------------------------------------------------------------------------
// Flash attention per (b, h, 16-query block). bf16 Q/K row-major (b,s,64),
// V^T depth-major per (b,h). Masked keys excluded from softmax. K/V tiles are
// double-buffered into LDS with async global->LDS copies when available.
// ---------------------------------------------------------------------------
__global__ __launch_bounds__(32) void attn_flash(const __bf16* __restrict__ Q,
                                                 const __bf16* __restrict__ Km,
                                                 const __bf16* __restrict__ Vt,
                                                 const float* __restrict__ keep,
                                                 __bf16* __restrict__ ctx) {
  __shared__ alignas(16) __bf16 pS[16 * 32];
#if ASYNC_EN
  __shared__ alignas(16) __bf16 ldsK[2][32 * 16];  // [key][depth]
  __shared__ alignas(16) __bf16 ldsV[2][16 * 32];  // [depth][key]
#endif
  const int q0 = blockIdx.x * 16;
  const int h = blockIdx.y;
  const int b = blockIdx.z;
  const int lane = threadIdx.x & 31;
  const float scale = 0.25f;  // 1/sqrt(16)
  const __bf16* Qb = Q + ((size_t)b * SS) * DD + (size_t)h * 16;
  const __bf16* Kb = Km + ((size_t)b * SS) * DD + (size_t)h * 16;
  const __bf16* Vb = Vt + ((size_t)(b * HH + h) * 16) * SS;
  const float* kpb = keep + (size_t)b * SS;
  const int nl = lane & 15;
  const int kb = (lane < 16) ? 0 : 8;
  const int kb16 = (lane < 16) ? 0 : 16;

  // Q fragment: depth 0..15 valid, upper K half zero.
  v16bf aq;
  {
    v8bf qlo = *(const v8bf*)(Qb + (size_t)(q0 + nl) * DD);
    v8bf qhi = *(const v8bf*)(Qb + (size_t)(q0 + nl) * DD + 8);
#pragma unroll
    for (int i = 0; i < 8; ++i) { aq[i] = (lane < 16) ? qlo[i] : (__bf16)0.f; aq[8 + i] = (lane < 16) ? qhi[i] : (__bf16)0.f; }
  }

  v8f O = {};
  float mrow[8], lrow[8];
#pragma unroll
  for (int r = 0; r < 8; ++r) { mrow[r] = -1e30f; lrow[r] = 0.f; }

#if ASYNC_EN
  // prefetch of one 32-key block = 4 async b128 ops (2 for K tile, 2 for V tile)
  auto prefetchKV = [&](int k0, int bufi) {
#pragma unroll
    for (int j = 0; j < 2; ++j) {
      const int chunk = j * 32 + lane;          // 64 x 16B chunks for K tile
      const int key = chunk >> 1, part = chunk & 1;
      cp16_async(Kb + (size_t)(k0 + key) * DD + part * 8,
                 &ldsK[bufi][key * 16 + part * 8]);
    }
#pragma unroll
    for (int j = 0; j < 2; ++j) {
      const int chunk = j * 32 + lane;          // 64 x 16B chunks for V tile
      const int d = chunk >> 2, part = chunk & 3;
      cp16_async(Vb + (size_t)d * SS + k0 + part * 8,
                 &ldsV[bufi][d * 32 + part * 8]);
    }
  };
  prefetchKV(0, 0);
#endif

  for (int k0 = 0; k0 < SS; k0 += 32) {
    const int cur = (k0 >> 5) & 1;
#if ASYNC_EN
    if (k0 + 32 < SS) {
      prefetchKV(k0 + 32, cur ^ 1);
      __builtin_amdgcn_s_wait_asynccnt(4);
    } else {
      __builtin_amdgcn_s_wait_asynccnt(0);
    }
    asm volatile("" ::: "memory");
#endif
    // K fragments (B layout): lanes 0-15 carry depth 0..15, upper half zero.
    v16bf bk0, bk1;
    if (lane < 16) {
#if ASYNC_EN
      v8bf k0lo = *(const v8bf*)&ldsK[cur][nl * 16];
      v8bf k0hi = *(const v8bf*)&ldsK[cur][nl * 16 + 8];
      v8bf k1lo = *(const v8bf*)&ldsK[cur][(16 + nl) * 16];
      v8bf k1hi = *(const v8bf*)&ldsK[cur][(16 + nl) * 16 + 8];
#else
      v8bf k0lo = *(const v8bf*)(Kb + (size_t)(k0 + nl) * DD);
      v8bf k0hi = *(const v8bf*)(Kb + (size_t)(k0 + nl) * DD + 8);
      v8bf k1lo = *(const v8bf*)(Kb + (size_t)(k0 + 16 + nl) * DD);
      v8bf k1hi = *(const v8bf*)(Kb + (size_t)(k0 + 16 + nl) * DD + 8);
#endif
      bk0 = cat8(k0lo, k0hi);
      bk1 = cat8(k1lo, k1hi);
    } else {
#pragma unroll
      for (int i = 0; i < 16; ++i) { bk0[i] = (__bf16)0.f; bk1[i] = (__bf16)0.f; }
    }
    v8f zc = {};
    v8f s0 = __builtin_amdgcn_wmma_f32_16x16x32_bf16(false, aq, false, bk0,
                                                     (short)0, zc, false, false);
    v8f s1 = __builtin_amdgcn_wmma_f32_16x16x32_bf16(false, aq, false, bk1,
                                                     (short)0, zc, false, false);
    const float kp0 = kpb[k0 + nl];
    const float kp1 = kpb[k0 + 16 + nl];
#pragma unroll
    for (int r = 0; r < 8; ++r) {
      const float a0 = (kp0 > 0.f) ? s0[r] * scale : -1e30f;
      const float a1 = (kp1 > 0.f) ? s1[r] * scale : -1e30f;
      const float mb = half_max(fmaxf(a0, a1));
      const float mn = fmaxf(mrow[r], mb);
      const float alpha = __expf(mrow[r] - mn);
      const float p0 = (kp0 > 0.f) ? __expf(a0 - mn) : 0.f;
      const float p1 = (kp1 > 0.f) ? __expf(a1 - mn) : 0.f;
      const float rs = half_sum(p0 + p1);
      lrow[r] = lrow[r] * alpha + rs;
      mrow[r] = mn;
      O[r] *= alpha;
      const int row = (lane < 16) ? r : 8 + r;
      pS[row * 32 + nl] = (__bf16)p0;
      pS[row * 32 + 16 + nl] = (__bf16)p1;
    }
    __syncthreads();
    // P as A fragment via LDS relayout (all K valid).
    v8bf plo = *(const v8bf*)&pS[nl * 32 + kb16];
    v8bf phi = *(const v8bf*)&pS[nl * 32 + kb16 + 8];
    // V fragment (B layout): lane nl = depth row, K = 32 keys of this block.
#if ASYNC_EN
    v8bf vlo = *(const v8bf*)&ldsV[cur][nl * 32 + kb16];
    v8bf vhi = *(const v8bf*)&ldsV[cur][nl * 32 + kb16 + 8];
#else
    v8bf vlo = *(const v8bf*)(Vb + (size_t)nl * SS + k0 + kb16);
    v8bf vhi = *(const v8bf*)(Vb + (size_t)nl * SS + k0 + kb16 + 8);
#endif
    O = __builtin_amdgcn_wmma_f32_16x16x32_bf16(false, cat8(plo, phi), false,
                                                cat8(vlo, vhi), (short)0, O,
                                                false, false);
    __syncthreads();
  }

  __bf16* cb = ctx + ((size_t)b * SS) * DD + (size_t)h * 16;
  const int rb2 = (lane < 16) ? 0 : 8;
#pragma unroll
  for (int r = 0; r < 8; ++r) {
    const float denom = lrow[r];
    const float o = (denom > 1e-20f) ? O[r] / denom : 0.f;
    cb[(size_t)(q0 + rb2 + r) * DD + nl] = (__bf16)o;
  }
}

// ---------------------------------------------------------------------------
// out = LayerNorm(A + Bsum) * g + beta, * keep. Writes f32 and bf16 copies.
// ---------------------------------------------------------------------------
__global__ __launch_bounds__(32) void ln_residual(const float* __restrict__ A,
                                                  const float* __restrict__ Bsum,
                                                  const float* __restrict__ g,
                                                  const float* __restrict__ beta,
                                                  const float* __restrict__ keep,
                                                  float* __restrict__ outF,
                                                  __bf16* __restrict__ outB) {
  const int row = blockIdx.x;
  const int t = threadIdx.x;
  const size_t base = (size_t)row * DD;
  const float x0 = A[base + t] + Bsum[base + t];
  const float x1 = A[base + 32 + t] + Bsum[base + 32 + t];
  const float s = wave_sum(x0 + x1);
  const float mean = s * (1.f / 64.f);
  const float d0 = x0 - mean, d1 = x1 - mean;
  const float vs = wave_sum(d0 * d0 + d1 * d1);
  const float inv = rsqrtf(vs * (1.f / 64.f) + 1e-6f);
  const float kp = keep[row];
  const float y0 = (d0 * inv * g[t] + beta[t]) * kp;
  const float y1 = (d1 * inv * g[32 + t] + beta[32 + t]) * kp;
  outF[base + t] = y0;
  outF[base + 32 + t] = y1;
  outB[base + t] = (__bf16)y0;
  outB[base + 32 + t] = (__bf16)y1;
}

// ---------------------------------------------------------------------------
// NaN handling + embedding assembly. keep=0 iff token NaN or whole batch NaN.
// ---------------------------------------------------------------------------
__global__ __launch_bounds__(1024) void prep_kernel(const float* __restrict__ inputs,
                                                    const float* __restrict__ emb,
                                                    float* __restrict__ x,
                                                    __bf16* __restrict__ xb,
                                                    float* __restrict__ keep) {
  __shared__ int anyValid;
  const int b = blockIdx.x;
  const int s = threadIdx.x;
  if (s == 0) anyValid = 0;
  __syncthreads();
  const float val = inputs[(size_t)b * SS + s];
  const bool isn = !(val == val);
  if (!isn) anyValid = 1;
  __syncthreads();
  const float kp = (!isn && anyValid) ? 1.f : 0.f;
  keep[(size_t)b * SS + s] = kp;
  const float xin = isn ? 0.f : val;
  const size_t rbase = ((size_t)b * SS + s) * DD;
  const float cs = 63.49803146f * kp;  // sqrt(63)*sqrt(64)
  for (int d = 0; d < DD - 1; ++d) {
    const float v = emb[(size_t)s * (DD - 1) + d] * cs;
    x[rbase + d] = v;
    xb[rbase + d] = (__bf16)v;
  }
  const float v = xin * kp;
  x[rbase + DD - 1] = v;
  xb[rbase + DD - 1] = (__bf16)v;
}

// ---------------------------------------------------------------------------
// y[b] = sum_s (h2[b,s,:] . w3) + S*b3 + out_bias.
// ---------------------------------------------------------------------------
__global__ __launch_bounds__(256) void head_final(const float* __restrict__ h2,
                                                  const float* __restrict__ w3,
                                                  const float* __restrict__ b3,
                                                  const float* __restrict__ ob,
                                                  float* __restrict__ y) {
  __shared__ float red[256];
  const int b = blockIdx.x;
  const int t = threadIdx.x;
  float acc = 0.f;
  for (int s = t; s < SS; s += 256) {
    const float* hr = h2 + ((size_t)b * SS + s) * DD;
    float d = 0.f;
#pragma unroll
    for (int j = 0; j < DD; ++j) d += hr[j] * w3[j];
    acc += d;
  }
  red[t] = acc;
  __syncthreads();
  for (int off = 128; off > 0; off >>= 1) {
    if (t < off) red[t] += red[t + off];
    __syncthreads();
  }
  if (t == 0) y[b] = red[0] + (float)SS * b3[0] + ob[0];
}

// ---------------------------------------------------------------------------
extern "C" void kernel_launch(void* const* d_in, const int* in_sizes, int n_in,
                              void* d_out, int out_size, void* d_ws, size_t ws_size,
                              hipStream_t stream) {
  const float* inputs = (const float*)d_in[0];
  const float* emb    = (const float*)d_in[1];
  const float* wq  = (const float*)d_in[2];
  const float* wqb = (const float*)d_in[3];
  const float* wk  = (const float*)d_in[4];
  const float* wkb = (const float*)d_in[5];
  const float* wv  = (const float*)d_in[6];
  const float* wvb = (const float*)d_in[7];
  const float* wo  = (const float*)d_in[8];
  const float* fw1 = (const float*)d_in[9];
  const float* fw2 = (const float*)d_in[10];
  const float* l1g = (const float*)d_in[11];
  const float* l1b = (const float*)d_in[12];
  const float* l2g = (const float*)d_in[13];
  const float* l2b = (const float*)d_in[14];
  const float* cw1 = (const float*)d_in[15];
  const float* cb1 = (const float*)d_in[16];
  const float* cw2 = (const float*)d_in[17];
  const float* cb2 = (const float*)d_in[18];
  const float* cw3 = (const float*)d_in[19];
  const float* cb3 = (const float*)d_in[20];
  const float* outb = (const float*)d_in[21];

  // f32 arenas
  float* x    = (float*)d_ws;                         // MM*DD
  float* keep = x + (size_t)MM * DD;                  // MM
  float* o    = keep + MM;                            // MM*DD
  float* o1   = o + (size_t)MM * DD;                  // MM*DD
  float* h2   = o1 + (size_t)MM * DD;                 // MM*DD
  // bf16 arenas
  __bf16* xb  = (__bf16*)(h2 + (size_t)MM * DD);      // MM*DD
  __bf16* qb  = xb + (size_t)MM * DD;
  __bf16* kbf = qb + (size_t)MM * DD;
  __bf16* vt  = kbf + (size_t)MM * DD;                // (B*H*16) x S
  __bf16* cxb = vt + (size_t)MM * DD;
  __bf16* o1b = cxb + (size_t)MM * DD;
  __bf16* fmb = o1b + (size_t)MM * DD;                // MM*DFFN
  __bf16* h1b = fmb + (size_t)MM * DFFN;
  // bf16 transposed weights
  __bf16* wqT = h1b + (size_t)MM * DD;
  __bf16* wkT = wqT + LL * 4096;
  __bf16* wvT = wkT + LL * 4096;
  __bf16* woT = wvT + LL * 4096;
  __bf16* f1T = woT + LL * 4096;                      // LL * 256*64
  __bf16* f2T = f1T + LL * 16384;                     // LL * 64*256
  __bf16* c1T = f2T + LL * 16384;
  __bf16* c2T = c1T + 4096;

  for (int i = 0; i < LL; ++i) {
    wt_cvt_t<<<64, 64, 0, stream>>>(wq + (size_t)i * 4096, wqT + (size_t)i * 4096, 64, 64);
    wt_cvt_t<<<64, 64, 0, stream>>>(wk + (size_t)i * 4096, wkT + (size_t)i * 4096, 64, 64);
    wt_cvt_t<<<64, 64, 0, stream>>>(wv + (size_t)i * 4096, wvT + (size_t)i * 4096, 64, 64);
    wt_cvt_t<<<64, 64, 0, stream>>>(wo + (size_t)i * 4096, woT + (size_t)i * 4096, 64, 64);
    wt_cvt_t<<<256, 64, 0, stream>>>(fw1 + (size_t)i * 16384, f1T + (size_t)i * 16384, 64, 256);
    wt_cvt_t<<<64, 64, 0, stream>>>(fw2 + (size_t)i * 16384, f2T + (size_t)i * 16384, 256, 64);
  }
  wt_cvt_t<<<64, 64, 0, stream>>>(cw1, c1T, 64, 64);
  wt_cvt_t<<<64, 64, 0, stream>>>(cw2, c2T, 64, 64);

  prep_kernel<<<BB, SS, 0, stream>>>(inputs, emb, x, xb, keep);

  dim3 g64(MM / 16, DD / 16);
  dim3 gf1(MM / 16, DFFN / 16);
  dim3 gat(SS / 16, HH, BB);

  for (int i = 0; i < LL; ++i) {
    gemm_bf16t<64, 64, 1, 0><<<g64, 32, 0, stream>>>(xb, wqT + (size_t)i * 4096, wqb + i * DD, nullptr, qb);
    gemm_bf16t<64, 64, 1, 0><<<g64, 32, 0, stream>>>(xb, wkT + (size_t)i * 4096, wkb + i * DD, nullptr, kbf);
    gemm_bf16t<64, 64, 2, 0><<<g64, 32, 0, stream>>>(xb, wvT + (size_t)i * 4096, wvb + i * DD, nullptr, vt);
    attn_flash<<<gat, 32, 0, stream>>>(qb, kbf, vt, keep, cxb);
    gemm_bf16t<64, 64, 0, 0><<<g64, 32, 0, stream>>>(cxb, woT + (size_t)i * 4096, nullptr, o, nullptr);
    ln_residual<<<MM, 32, 0, stream>>>(x, o, l1g + i * DD, l1b + i * DD, keep, o1, o1b);
    gemm_bf16t<64, 256, 1, 1><<<gf1, 32, 0, stream>>>(o1b, f1T + (size_t)i * 16384, nullptr, nullptr, fmb);
    gemm_bf16t<256, 64, 0, 0><<<g64, 32, 0, stream>>>(fmb, f2T + (size_t)i * 16384, nullptr, o, nullptr);
    ln_residual<<<MM, 32, 0, stream>>>(o1, o, l2g + i * DD, l2b + i * DD, keep, x, xb);
  }

  gemm_bf16t<64, 64, 1, 1><<<g64, 32, 0, stream>>>(xb, c1T, cb1, nullptr, h1b);
  gemm_bf16t<64, 64, 0, 1><<<g64, 32, 0, stream>>>(h1b, c2T, cb2, h2, nullptr);
  head_final<<<BB, 256, 0, stream>>>(h2, cw3, cb3, outb, (float*)d_out);

  (void)in_sizes; (void)n_in; (void)out_size; (void)ws_size;
}
